// AttRegressor_60387240182515
// MI455X (gfx1250) — matrix operand order repaired and checked by reading, not compiled, not run on hardware
//
#include <hip/hip_runtime.h>
#include <math.h>

// ---------------------------------------------------------------------------
// Types for CDNA5 WMMA (wave32, v_wmma_f32_16x16x32_bf16)
// ---------------------------------------------------------------------------
typedef __bf16 bf16_t;
typedef bf16_t v16bf __attribute__((ext_vector_type(16)));
typedef bf16_t v8bf  __attribute__((ext_vector_type(8)));
typedef float  v8f   __attribute__((ext_vector_type(8)));

static constexpr int Bn = 4096, Dn = 128, Kn = 4096, En = 1024, Hn = 2048, Vn = 64;

__device__ __forceinline__ bf16_t f2bf(float f) {
  unsigned u = __float_as_uint(f);
  u += 0x7FFFu + ((u >> 16) & 1u);        // round-to-nearest-even
  unsigned short h = (unsigned short)(u >> 16);
  return __builtin_bit_cast(bf16_t, h);
}

// A fragment, 16x32 bf16, row-major A[M][Kd].
// Lanes 0-15: row M=lane, V0-3 hold K=0..7, V4-7 hold K=8..15.
// Lanes 16-31: row M=lane-16, V0-3 hold K=16..23, V4-7 hold K=24..31.
__device__ __forceinline__ v16bf load_a_frag(const bf16_t* __restrict__ p) {
  v8bf lo = *(const v8bf*)p;          // K .. +8
  v8bf hi = *(const v8bf*)(p + 8);    // K+8 .. +8
  return __builtin_shufflevector(lo, hi, 0,1,2,3,4,5,6,7,8,9,10,11,12,13,14,15);
}

__device__ __forceinline__ v8f wmma_bf16(v16bf a, v16bf b, v8f c) {
  return __builtin_amdgcn_wmma_f32_16x16x32_bf16(false, a, false, b, (short)0, c,
                                                 false, false);
}

// Fragment sets for register double-buffering.
struct Frag42 { v16bf a[4]; v16bf b[2]; };
struct Frag22 { v16bf a[2]; v16bf b[2]; };

template <int Kd>
__device__ __forceinline__ void load42(Frag42& f, const bf16_t* __restrict__ pa,
                                       const bf16_t* __restrict__ pb, int k0) {
#pragma unroll
  for (int i = 0; i < 4; i++) f.a[i] = load_a_frag(pa + (size_t)(16 * i) * Kd + k0);
#pragma unroll
  for (int j = 0; j < 2; j++) f.b[j] = *(const v16bf*)(pb + (size_t)(16 * j) * Kd + k0);
}

__device__ __forceinline__ void mma42(v8f (&acc)[4][2], const Frag42& f) {
#pragma unroll
  for (int i = 0; i < 4; i++)
#pragma unroll
    for (int j = 0; j < 2; j++) acc[i][j] = wmma_bf16(f.a[i], f.b[j], acc[i][j]);
}

template <int Kd>
__device__ __forceinline__ void load22(Frag22& f, const bf16_t* __restrict__ pa,
                                       const bf16_t* __restrict__ pb, int k0) {
#pragma unroll
  for (int i = 0; i < 2; i++) f.a[i] = load_a_frag(pa + (size_t)(16 * i) * Kd + k0);
#pragma unroll
  for (int j = 0; j < 2; j++) f.b[j] = *(const v16bf*)(pb + (size_t)(16 * j) * Kd + k0);
}

__device__ __forceinline__ void mma22(v8f (&acc)[2][2], const Frag22& f) {
#pragma unroll
  for (int i = 0; i < 2; i++)
#pragma unroll
    for (int j = 0; j < 2; j++) acc[i][j] = wmma_bf16(f.a[i], f.b[j], acc[i][j]);
}

// ---------------------------------------------------------------------------
// Main BF16 WMMA GEMM: C[M][N] = A[M][Kd] * Bt[N][Kd]^T (+bias, +relu)
// Block tile 128x128, 256 threads = 8 waves, each wave a 64x32 patch (4x2 tiles).
// Register double-buffered over K steps of 64 (Kd % 64 == 0).
// EPI: 0 = f32 out, no bias; 1 = bf16 out, bias+relu; 2 = f32 out, bias.
// ---------------------------------------------------------------------------
template <int EPI, int M, int N, int Kd>
__global__ void __launch_bounds__(256)
gemm128x128(const bf16_t* __restrict__ A, const bf16_t* __restrict__ Bt,
            const float* __restrict__ bias, void* __restrict__ Cout) {
  static_assert(Kd % 64 == 0, "K must be a multiple of 64");
  const int lane = threadIdx.x & 31;
  const int wave = threadIdx.x >> 5;
  const int wm = blockIdx.y * 128 + (wave & 1) * 64;
  const int wn = blockIdx.x * 128 + (wave >> 1) * 32;
  const int half = lane >> 4;
  const int l15 = lane & 15;

  const bf16_t* pa = A  + (size_t)(wm + l15) * Kd + half * 16;  // +16*i*Kd, +k0
  const bf16_t* pb = Bt + (size_t)(wn + l15) * Kd + half * 16;  // +16*j*Kd, +k0

  v8f acc[4][2];
#pragma unroll
  for (int i = 0; i < 4; i++)
#pragma unroll
    for (int j = 0; j < 2; j++)
#pragma unroll
      for (int v = 0; v < 8; v++) acc[i][j][v] = 0.0f;

  Frag42 f0, f1;
  load42<Kd>(f0, pa, pb, 0);
#pragma unroll 1
  for (int k0 = 0; k0 < Kd; k0 += 64) {
    load42<Kd>(f1, pa, pb, k0 + 32);     // in flight while f0's WMMAs run
    mma42(acc, f0);
    if (k0 + 64 < Kd) {
      __builtin_prefetch(pa + k0 + 64, 0, 3);   // global_prefetch_b8
      __builtin_prefetch(pb + k0 + 64, 0, 3);
      load42<Kd>(f0, pa, pb, k0 + 64);   // in flight while f1's WMMAs run
    }
    mma42(acc, f1);
  }

  // C layout: VGPR v -> row v (lanes 0-15) or v+8 (lanes 16-31); lane&15 -> col.
  const size_t cbase = (size_t)(wm + half * 8) * N + wn + l15;
#pragma unroll
  for (int j = 0; j < 2; j++) {
    const float bv = (EPI != 0) ? bias[wn + 16 * j + l15] : 0.0f;
#pragma unroll
    for (int i = 0; i < 4; i++) {
#pragma unroll
      for (int v = 0; v < 8; v++) {
        const size_t idx = cbase + (size_t)(16 * i + v) * N + 16 * j;
        float x = acc[i][j][v] + bv;
        if (EPI == 1) {
          x = fmaxf(x, 0.0f);
          ((bf16_t*)Cout)[idx] = f2bf(x);
        } else {
          ((float*)Cout)[idx] = x;
        }
      }
    }
  }
}

// Skinny GEMM for att @ values: tile 128x64, each wave 32x32 (2x2 tiles), f32 out.
template <int M, int N, int Kd>
__global__ void __launch_bounds__(256)
gemm128x64(const bf16_t* __restrict__ A, const bf16_t* __restrict__ Bt,
           float* __restrict__ Cout) {
  static_assert(Kd % 64 == 0, "K must be a multiple of 64");
  const int lane = threadIdx.x & 31;
  const int wave = threadIdx.x >> 5;
  const int wm = blockIdx.y * 128 + (wave >> 1) * 32;
  const int wn = (wave & 1) * 32;
  const int half = lane >> 4;
  const int l15 = lane & 15;

  const bf16_t* pa = A  + (size_t)(wm + l15) * Kd + half * 16;
  const bf16_t* pb = Bt + (size_t)(wn + l15) * Kd + half * 16;

  v8f acc[2][2];
#pragma unroll
  for (int i = 0; i < 2; i++)
#pragma unroll
    for (int j = 0; j < 2; j++)
#pragma unroll
      for (int v = 0; v < 8; v++) acc[i][j][v] = 0.0f;

  Frag22 f0, f1;
  load22<Kd>(f0, pa, pb, 0);
#pragma unroll 1
  for (int k0 = 0; k0 < Kd; k0 += 64) {
    load22<Kd>(f1, pa, pb, k0 + 32);
    mma22(acc, f0);
    if (k0 + 64 < Kd) {
      __builtin_prefetch(pa + k0 + 64, 0, 3);
      __builtin_prefetch(pb + k0 + 64, 0, 3);
      load22<Kd>(f0, pa, pb, k0 + 64);
    }
    mma22(acc, f1);
  }

  const size_t cbase = (size_t)(wm + half * 8) * N + wn + l15;
#pragma unroll
  for (int j = 0; j < 2; j++)
#pragma unroll
    for (int i = 0; i < 2; i++)
#pragma unroll
      for (int v = 0; v < 8; v++)
        Cout[cbase + (size_t)(16 * i + v) * N + 16 * j] = acc[i][j][v];
}

// ---------------------------------------------------------------------------
// Data-prep kernels
// ---------------------------------------------------------------------------
__global__ void convert_to_bf16(const float* __restrict__ in, bf16_t* __restrict__ out, int n) {
  for (int i = blockIdx.x * blockDim.x + threadIdx.x; i < n; i += gridDim.x * blockDim.x)
    out[i] = f2bf(in[i]);
}

// out[c][r] = bf16(in[r][c]); R,C multiples of 32. Block (32,8), LDS 32x33 tile.
__global__ void __launch_bounds__(256)
transpose_to_bf16(const float* __restrict__ in, bf16_t* __restrict__ out, int R, int C) {
  __shared__ float tile[32][33];
  const int tx = threadIdx.x, ty = threadIdx.y;
  const int r0 = blockIdx.y * 32, c0 = blockIdx.x * 32;
#pragma unroll
  for (int i = 0; i < 4; i++)
    tile[ty + i * 8][tx] = in[(size_t)(r0 + ty + i * 8) * C + (c0 + tx)];
  __syncthreads();
#pragma unroll
  for (int i = 0; i < 4; i++)
    out[(size_t)(c0 + ty + i * 8) * R + (r0 + tx)] = f2bf(tile[tx][ty + i * 8]);
}

__global__ void rownorm2(const float* __restrict__ in, float* __restrict__ out,
                         int rows, int cols) {
  int r = blockIdx.x * blockDim.x + threadIdx.x;
  if (r >= rows) return;
  const float* p = in + (size_t)r * cols;
  float s = 0.0f;
  for (int j = 0; j < cols; j++) s += p[j] * p[j];
  out[r] = s;
}

// ---------------------------------------------------------------------------
// Row ops (FP32 precision): distance -> score -> row-normalize ; softmax.
// One block per row, Kc = 4096 = 256 threads * 16.
// ---------------------------------------------------------------------------
__global__ void __launch_bounds__(256)
preprocess_rows(const float* __restrict__ g, const float* __restrict__ x2,
                const float* __restrict__ k2, bf16_t* __restrict__ s, int Kc) {
  const int row = blockIdx.x, tid = threadIdx.x;
  const float* grow = g + (size_t)row * Kc;
  const float xr = x2[row];
  const int per = Kc >> 8;  // 16
  float diff[16];
  float lmax = 0.0f;
  for (int t = 0; t < per; t++) {
    int j = tid + (t << 8);
    float d2 = xr + k2[j] - 2.0f * grow[j];
    float d = sqrtf(fmaxf(d2, 0.0f));
    diff[t] = d;
    lmax = fmaxf(lmax, d);
  }
  __shared__ float red[256];
  red[tid] = lmax; __syncthreads();
  for (int st = 128; st > 0; st >>= 1) {
    if (tid < st) red[tid] = fmaxf(red[tid], red[tid + st]);
    __syncthreads();
  }
  const float scale = -10.0f / red[0];
  __syncthreads();
  float sc[16];
  float lsum = 0.0f;
  for (int t = 0; t < per; t++) { sc[t] = __expf(diff[t] * scale); lsum += sc[t]; }
  red[tid] = lsum; __syncthreads();
  for (int st = 128; st > 0; st >>= 1) {
    if (tid < st) red[tid] += red[tid + st];
    __syncthreads();
  }
  const float inv = 1.0f / red[0];
  for (int t = 0; t < per; t++)
    s[(size_t)row * Kc + tid + (t << 8)] = f2bf(sc[t] * inv);
}

__global__ void __launch_bounds__(256)
softmax_rows(const float* __restrict__ logit, bf16_t* __restrict__ att, int Kc) {
  const int row = blockIdx.x, tid = threadIdx.x;
  const float* p = logit + (size_t)row * Kc;
  const int per = Kc >> 8;
  float v[16];
  float lmax = -3.402823e38f;
  for (int t = 0; t < per; t++) {
    v[t] = p[tid + (t << 8)];
    lmax = fmaxf(lmax, v[t]);
  }
  __shared__ float red[256];
  red[tid] = lmax; __syncthreads();
  for (int st = 128; st > 0; st >>= 1) {
    if (tid < st) red[tid] = fmaxf(red[tid], red[tid + st]);
    __syncthreads();
  }
  const float m = red[0];
  __syncthreads();
  float lsum = 0.0f;
  for (int t = 0; t < per; t++) { v[t] = __expf(v[t] - m); lsum += v[t]; }
  red[tid] = lsum; __syncthreads();
  for (int st = 128; st > 0; st >>= 1) {
    if (tid < st) red[tid] += red[tid + st];
    __syncthreads();
  }
  const float inv = 1.0f / red[0];
  for (int t = 0; t < per; t++)
    att[(size_t)row * Kc + tid + (t << 8)] = f2bf(v[t] * inv);
}

// ---------------------------------------------------------------------------
// Launch: full pipeline on `stream`; scratch carved from d_ws (~151 MB).
// ---------------------------------------------------------------------------
extern "C" void kernel_launch(void* const* d_in, const int* in_sizes, int n_in,
                              void* d_out, int out_size, void* d_ws, size_t ws_size,
                              hipStream_t stream) {
  (void)in_sizes; (void)n_in; (void)out_size; (void)ws_size;
  const float* x        = (const float*)d_in[0];
  const float* keys     = (const float*)d_in[1];
  const float* values   = (const float*)d_in[2];
  const float* W_embed  = (const float*)d_in[3];
  const float* b_embed  = (const float*)d_in[4];
  const float* W_hidden = (const float*)d_in[5];
  const float* b_hidden = (const float*)d_in[6];
  const float* W_att    = (const float*)d_in[7];
  const float* b_att    = (const float*)d_in[8];
  float* out = (float*)d_out;

  char* w = (char*)d_ws;
  size_t off = 0;
  auto take = [&](size_t bytes) -> char* {
    char* p = w + off;
    off = (off + bytes + 255) & ~(size_t)255;
    return p;
  };
  bf16_t* xb    = (bf16_t*)take((size_t)Bn * Dn * 2);   // x in bf16
  bf16_t* keysb = (bf16_t*)take((size_t)Kn * Dn * 2);   // keys bf16 ([N][K] already)
  bf16_t* WembT = (bf16_t*)take((size_t)En * Kn * 2);   // W_embed^T
  bf16_t* WhidT = (bf16_t*)take((size_t)Hn * En * 2);   // W_hidden^T
  bf16_t* WattT = (bf16_t*)take((size_t)Kn * Hn * 2);   // W_att^T
  bf16_t* valT  = (bf16_t*)take((size_t)Vn * Kn * 2);   // values^T
  float*  x2    = (float*)take((size_t)Bn * 4);
  float*  k2    = (float*)take((size_t)Kn * 4);
  float*  fbuf  = (float*)take((size_t)Bn * Kn * 4);    // x@keys^T, then logits
  bf16_t* sb    = (bf16_t*)take((size_t)Bn * Kn * 2);   // scores, then attention
  bf16_t* eb    = (bf16_t*)take((size_t)Bn * En * 2);
  bf16_t* hb    = (bf16_t*)take((size_t)Bn * Hn * 2);

  dim3 tblk(32, 8);
  convert_to_bf16<<<512, 256, 0, stream>>>(x, xb, Bn * Dn);
  convert_to_bf16<<<512, 256, 0, stream>>>(keys, keysb, Kn * Dn);
  transpose_to_bf16<<<dim3(En / 32, Kn / 32), tblk, 0, stream>>>(W_embed, WembT, Kn, En);
  transpose_to_bf16<<<dim3(Hn / 32, En / 32), tblk, 0, stream>>>(W_hidden, WhidT, En, Hn);
  transpose_to_bf16<<<dim3(Kn / 32, Hn / 32), tblk, 0, stream>>>(W_att, WattT, Hn, Kn);
  transpose_to_bf16<<<dim3(Vn / 32, Kn / 32), tblk, 0, stream>>>(values, valT, Kn, Vn);
  rownorm2<<<Bn / 256, 256, 0, stream>>>(x, x2, Bn, Dn);
  rownorm2<<<Kn / 256, 256, 0, stream>>>(keys, k2, Kn, Dn);

  // g = x @ keys^T                                    [B,K]
  gemm128x128<0, Bn, Kn, Dn><<<dim3(Kn / 128, Bn / 128), 256, 0, stream>>>(
      xb, keysb, nullptr, fbuf);
  // s = rownorm(exp(-10*dist/max))                    [B,K] bf16
  preprocess_rows<<<Bn, 256, 0, stream>>>(fbuf, x2, k2, sb, Kn);
  // e = relu(s @ W_embed + b_embed)                   [B,E] bf16
  gemm128x128<1, Bn, En, Kn><<<dim3(En / 128, Bn / 128), 256, 0, stream>>>(
      sb, WembT, b_embed, eb);
  // h = relu(e @ W_hidden + b_hidden)                 [B,H] bf16
  gemm128x128<1, Bn, Hn, En><<<dim3(Hn / 128, Bn / 128), 256, 0, stream>>>(
      eb, WhidT, b_hidden, hb);
  // logits = h @ W_att + b_att                        [B,K] f32
  gemm128x128<2, Bn, Kn, Hn><<<dim3(Kn / 128, Bn / 128), 256, 0, stream>>>(
      hb, WattT, b_att, fbuf);
  // att = softmax(logits)                             [B,K] bf16
  softmax_rows<<<Bn, 256, 0, stream>>>(fbuf, sb, Kn);
  // out = att @ values                                [B,V] f32
  gemm128x64<Bn, Vn, Kn><<<dim3(1, Bn / 128), 256, 0, stream>>>(sb, valT, out);
}
